// StandardAttention_39865886441709
// MI455X (gfx1250) — matrix operand order repaired and checked by reading, not compiled, not run on hardware
//
#include <hip/hip_runtime.h>
#include <hip/hip_bf16.h>

// ---------------------------------------------------------------------------
// Problem constants (from reference): B=2, L=2048, D=2048, H=16, HD=128
// ---------------------------------------------------------------------------
constexpr int Bc  = 2;
constexpr int Lc  = 2048;
constexpr int Dc  = 2048;
constexpr int Hc  = 16;
constexpr int HDc = 128;
constexpr int HD2c = 64;

typedef __bf16 bf16_t;
typedef __attribute__((ext_vector_type(16))) __bf16 bf16x16;
typedef __attribute__((ext_vector_type(8)))  __bf16 bf16x8;
typedef __attribute__((ext_vector_type(4)))  __bf16 bf16x4;
typedef __attribute__((ext_vector_type(8)))  float  v8f;
typedef __attribute__((ext_vector_type(4)))  unsigned int u32x4;
typedef __attribute__((ext_vector_type(8)))  int i32x8;
typedef __attribute__((ext_vector_type(4)))  int i32x4;

// ---------------------------------------------------------------------------
// WMMA fragment helpers (16x16x32 bf16; A/B use this code's contiguous k order, C/D per ISA 7.12.2)
// A (16xK=32): lane half h -> K[16h..16h+15] (contiguous 16 per lane)
// B (K=32x16): lane n = lane&15, k = 16*half + e (contiguous 16 per lane)
// C/D (16x16 f32): vgpr v -> row = v + 8*half, col = lane&15
// ---------------------------------------------------------------------------
__device__ __forceinline__ bf16x16 load_a_frag(const bf16_t* __restrict__ base,
                                               int row, int ld, int kbase, int lane) {
  const int half = (lane >> 4) & 1;
  const bf16_t* p = base + (size_t)row * ld + kbase;
  union { bf16x16 v; bf16x8 h[2]; } u;
  u.h[0] = *(const bf16x8*)(p + half * 16);
  u.h[1] = *(const bf16x8*)(p + half * 16 + 8);
  return u.v;
}

// B[k][n] = M[n][k] for row-major M (n x ld): per-lane 16 contiguous elements.
__device__ __forceinline__ bf16x16 load_b_frag(const bf16_t* __restrict__ base,
                                               int col, int ld, int kbase, int lane) {
  const int half = (lane >> 4) & 1;
  return *(const bf16x16*)(base + (size_t)col * ld + kbase + half * 16);
}

__device__ __forceinline__ v8f wmma_bf16(bf16x16 a, bf16x16 b, v8f c) {
  return __builtin_amdgcn_wmma_f32_16x16x32_bf16(false, a, false, b, (short)0, c,
                                                 false, false);
}

// ---------------------------------------------------------------------------
// Tensor Data Mover: 2D bf16 tile (tile_d0 contiguous elems x tile_d1 rows)
// global -> LDS. Descriptor per CDNA5 ISA.
// This toolchain exposes the 6-arg builtin:
//   (uint32x4 g0, int32x8 g1, int32x4 g2, int32x4 g3, int32x8 pad, i32 cpol)
// ---------------------------------------------------------------------------
#if __has_builtin(__builtin_amdgcn_tensor_load_to_lds)
#define HAVE_TDM 1
#else
#define HAVE_TDM 0
#endif

#if HAVE_TDM
__device__ __forceinline__ void tdm_load_2d_bf16(unsigned lds_off, const void* gaddr,
                                                 unsigned tile_d0, unsigned tile_d1,
                                                 unsigned tensor_d0, unsigned tensor_d1,
                                                 unsigned stride_d0 /*elements*/) {
  const unsigned long long ga = (unsigned long long)gaddr;
  u32x4 g0;
  g0[0] = 1u;                                           // count=1, user descriptor
  g0[1] = lds_off;                                      // lds_addr (bytes)
  g0[2] = (unsigned)(ga & 0xffffffffu);                 // global_addr[31:0]
  g0[3] = (unsigned)((ga >> 32) & 0x01ffffffu)          // global_addr[56:32]
        | (2u << 30);                                   // type=2 ("image")
  i32x8 g1;
  g1[0] = (int)(1u << 16);                              // data_size=1 (2B), wg_mask=0
  g1[1] = (int)((tensor_d0 & 0xffffu) << 16);           // tensor_dim0[15:0] @ [63:48]
  g1[2] = (int)(((tensor_d0 >> 16) & 0xffffu)           // tensor_dim0[31:16]
        | ((tensor_d1 & 0xffffu) << 16));               // tensor_dim1[15:0]
  g1[3] = (int)(((tensor_d1 >> 16) & 0xffffu)           // tensor_dim1[31:16]
        | ((tile_d0 & 0xffffu) << 16));                 // tile_dim0
  g1[4] = (int)(tile_d1 & 0xffffu);                     // tile_dim1 (tile_dim2=0)
  g1[5] = (int)stride_d0;                               // tensor_dim0_stride[31:0]
  g1[6] = 0;                                            // stride hi / dim1_stride
  g1[7] = 0;
  const i32x4 z4 = {0, 0, 0, 0};
  const i32x8 z8 = {0, 0, 0, 0, 0, 0, 0, 0};
  __builtin_amdgcn_tensor_load_to_lds(g0, g1, z4, z4, z8, 0);
}
#endif

// ---------------------------------------------------------------------------
// Kernel 0: f32 -> bf16 conversion (x4 vectorized)
// ---------------------------------------------------------------------------
__global__ void cvt_f32_bf16_v4(const float4* __restrict__ in,
                                bf16x4* __restrict__ out, int n4) {
  int i = blockIdx.x * blockDim.x + threadIdx.x;
  if (i < n4) {
    float4 f = in[i];
    bf16x4 o = {(bf16_t)f.x, (bf16_t)f.y, (bf16_t)f.z, (bf16_t)f.w};
    out[i] = o;
  }
}

// ---------------------------------------------------------------------------
// Kernel 1: fused QKV projection + RoPE, 2x2 register blocking per wave.
// Each wave: 32 rows x 32 cols of Q, K and V simultaneously.
// Per k-step: 2 A frags reused over 3 weights x 2 col tiles (12 WMMAs).
// Q,K stored bf16 (B,H,L,HD) with RoPE applied; V stored bf16 (B,H,HD,L).
// ---------------------------------------------------------------------------
__global__ __launch_bounds__(256) void qkv_rope_kernel(
    const bf16_t* __restrict__ xb,
    const bf16_t* __restrict__ wq, const bf16_t* __restrict__ wk,
    const bf16_t* __restrict__ wv,
    const float* __restrict__ fcos, const float* __restrict__ fsin,
    bf16_t* __restrict__ Qb, bf16_t* __restrict__ Kb, bf16_t* __restrict__ Vt) {
  const int lane = threadIdx.x & 31;
  const int wave = threadIdx.x >> 5;
  const int blk  = blockIdx.x * 8 + wave;      // one 32x32 region per wave
  const int col_blocks = Dc / 32;              // 64
  const int rb = blk / col_blocks;             // over B*L/32 = 128
  const int cb = blk - rb * col_blocks;
  const int row0 = rb * 32;
  const int col0 = cb * 32;
  const int ln = lane & 15;
  const int hi = (lane >> 4) & 1;

  v8f cq[2][2] = {}, ck[2][2] = {}, cv[2][2] = {};

  for (int kk = 0; kk < Dc; kk += 32) {
    bf16x16 a0 = load_a_frag(xb, row0 + ln, Dc, kk, lane);
    bf16x16 a1 = load_a_frag(xb, row0 + 16 + ln, Dc, kk, lane);
#pragma unroll
    for (int ct = 0; ct < 2; ++ct) {
      const int bc = col0 + ct * 16 + ln;
      bf16x16 bq = load_b_frag(wq, bc, Dc, kk, lane);
      cq[0][ct] = wmma_bf16(a0, bq, cq[0][ct]);
      cq[1][ct] = wmma_bf16(a1, bq, cq[1][ct]);
      bf16x16 bk = load_b_frag(wk, bc, Dc, kk, lane);
      ck[0][ct] = wmma_bf16(a0, bk, ck[0][ct]);
      ck[1][ct] = wmma_bf16(a1, bk, ck[1][ct]);
      bf16x16 bv = load_b_frag(wv, bc, Dc, kk, lane);
      cv[0][ct] = wmma_bf16(a0, bv, cv[0][ct]);
      cv[1][ct] = wmma_bf16(a1, bv, cv[1][ct]);
    }
  }

  // RoPE on Q,K accumulators + scatter stores.
  const int b = row0 / Lc;                     // 32-row block never straddles batches
#pragma unroll
  for (int ct = 0; ct < 2; ++ct) {
    const int f     = col0 + ct * 16 + ln;     // feature index for this lane/tile
    const int h     = f >> 7;
    const int hd    = f & (HDc - 1);
    const int pairi = hd >> 1;
    const float sgn = (f & 1) ? 1.0f : -1.0f;  // r0 = t0*c - t1*s ; r1 = t1*c + t0*s
#pragma unroll
    for (int rt = 0; rt < 2; ++rt) {
#pragma unroll
      for (int v = 0; v < 8; ++v) {
        const int m = rt * 16 + v + (hi << 3);
        const int l = (row0 + m) & (Lc - 1);
        const float c = fcos[l * HD2c + pairi];
        const float s = fsin[l * HD2c + pairi];
        const float q0 = cq[rt][ct][v], k0 = ck[rt][ct][v];
        const float qp = __shfl_xor(q0, 1, 32);  // partner feature f^1
        const float kp = __shfl_xor(k0, 1, 32);
        const float qn = q0 * c + sgn * qp * s;
        const float kn = k0 * c + sgn * kp * s;
        const size_t qaddr = ((size_t)(b * Hc + h) * Lc + l) * HDc + hd;
        Qb[qaddr] = (bf16_t)qn;
        Kb[qaddr] = (bf16_t)kn;
        const size_t vaddr = ((size_t)(b * Hc + h) * HDc + hd) * Lc + l;
        Vt[vaddr] = (bf16_t)cv[rt][ct][v];
      }
    }
  }
}

// ---------------------------------------------------------------------------
// Kernel 2: causal flash attention. One wave per (b,h, 16-query tile).
// Online softmax over kv in steps of 32; P relayouted C->A via LDS.
// Output written bf16 directly into (B,L,D) layout for the final projection.
// ---------------------------------------------------------------------------
#define ATT_WAVES 4
__global__ __launch_bounds__(32 * ATT_WAVES) void attn_kernel(
    const bf16_t* __restrict__ Qb, const bf16_t* __restrict__ Kb,
    const bf16_t* __restrict__ Vt, bf16_t* __restrict__ attn) {
  __shared__ bf16_t plds[ATT_WAVES][16 * 32];

  const int lane = threadIdx.x & 31;
  const int wave = threadIdx.x >> 5;
  const int wid  = blockIdx.x * ATT_WAVES + wave;  // 0 .. B*H*(L/16)-1
  const int qt   = wid & (Lc / 16 - 1);
  const int bh   = wid / (Lc / 16);
  const int b    = bh >> 4;
  const int h    = bh & (Hc - 1);
  const int hi   = (lane >> 4) & 1;
  const int ln   = lane & 15;

  const bf16_t* Qh = Qb + (size_t)bh * Lc * HDc;
  const bf16_t* Kh = Kb + (size_t)bh * Lc * HDc;
  const bf16_t* Vh = Vt + (size_t)bh * HDc * Lc;

  // Preload Q as 4 A-fragments covering HD=128.
  bf16x16 qf[4];
  const int qrow = qt * 16 + ln;
#pragma unroll
  for (int c = 0; c < 4; ++c) qf[c] = load_a_frag(Qh, qrow, HDc, c * 32, lane);

  v8f o[8];
  float mrow[8], lrow[8];
#pragma unroll
  for (int i = 0; i < 8; ++i) { o[i] = (v8f){}; mrow[i] = -3.0e38f; lrow[i] = 0.0f; }

  const float scale = 0.08838834764831845f;  // 1/sqrt(128)
  const int kv_end = qt * 16 + 16;

  for (int kv0 = 0; kv0 < kv_end; kv0 += 32) {
    // --- scores: two 16x16 tiles, K-dim = HD = 128 (4 chained WMMAs each) ---
    v8f s0 = {}, s1 = {};
#pragma unroll
    for (int c = 0; c < 4; ++c)
      s0 = wmma_bf16(qf[c], load_b_frag(Kh, kv0 + ln, HDc, c * 32, lane), s0);
#pragma unroll
    for (int c = 0; c < 4; ++c)
      s1 = wmma_bf16(qf[c], load_b_frag(Kh, kv0 + 16 + ln, HDc, c * 32, lane), s1);

    // --- scale, causal mask, online softmax (rows striped across 16 lanes) ---
#pragma unroll
    for (int v = 0; v < 8; ++v) {
      const int m  = v + (hi << 3);
      const int q  = qt * 16 + m;
      const int n0 = kv0 + ln;
      const int n1 = n0 + 16;
      float e0 = (n0 <= q) ? s0[v] * scale : -3.0e38f;
      float e1 = (n1 <= q) ? s1[v] * scale : -3.0e38f;
      float mx = fmaxf(e0, e1);
      mx = fmaxf(mx, __shfl_xor(mx, 1, 32));
      mx = fmaxf(mx, __shfl_xor(mx, 2, 32));
      mx = fmaxf(mx, __shfl_xor(mx, 4, 32));
      mx = fmaxf(mx, __shfl_xor(mx, 8, 32));
      const float mnew  = fmaxf(mrow[v], mx);
      const float alpha = __expf(mrow[v] - mnew);
      const float p0 = __expf(e0 - mnew);
      const float p1 = __expf(e1 - mnew);
      float ps = p0 + p1;
      ps += __shfl_xor(ps, 1, 32);
      ps += __shfl_xor(ps, 2, 32);
      ps += __shfl_xor(ps, 4, 32);
      ps += __shfl_xor(ps, 8, 32);
      lrow[v] = lrow[v] * alpha + ps;
      mrow[v] = mnew;
#pragma unroll
      for (int nt = 0; nt < 8; ++nt) o[nt][v] *= alpha;
      // Stage P (bf16) in LDS: row m, 32 kv columns.
      plds[wave][m * 32 + ln]      = (bf16_t)p0;
      plds[wave][m * 32 + 16 + ln] = (bf16_t)p1;
    }

    // --- re-read P in A-fragment layout (16x32) ---
    bf16x16 pa;
    {
      const bf16_t* p = &plds[wave][ln * 32];
      union { bf16x16 v; bf16x8 h[2]; } u;
      u.h[0] = *(const bf16x8*)(p + hi * 16);
      u.h[1] = *(const bf16x8*)(p + hi * 16 + 8);
      pa = u.v;
    }

    // --- o += P · V  (V stored transposed: B-fragment reads contiguous) ---
#pragma unroll
    for (int nt = 0; nt < 8; ++nt) {
      bf16x16 bv = load_b_frag(Vh, nt * 16 + ln, Lc, kv0, lane);
      o[nt] = wmma_bf16(pa, bv, o[nt]);
    }
  }

  // --- normalize and scatter into (B, L, D) bf16 for the output projection ---
#pragma unroll
  for (int nt = 0; nt < 8; ++nt) {
#pragma unroll
    for (int v = 0; v < 8; ++v) {
      const int m = v + (hi << 3);
      const int l = qt * 16 + m;
      const int f = h * HDc + nt * 16 + ln;
      attn[((size_t)b * Lc + l) * Dc + f] = (bf16_t)(o[nt][v] / lrow[v]);
    }
  }
}

// ---------------------------------------------------------------------------
// Kernel 3: output projection  out = attn @ Wo^T  (f32 output).
// Block = 8 waves covering 128 rows x 64 cols; each wave a 32x32 tile pair.
// Wo panel (64 cols x 32 k, bf16) staged into LDS by the Tensor Data Mover
// (wave 0 issues tensor_load_to_lds, waits TENSORcnt, workgroup barrier).
// ---------------------------------------------------------------------------
__global__ __launch_bounds__(256) void proj_kernel(
    const bf16_t* __restrict__ ab, const bf16_t* __restrict__ wo,
    float* __restrict__ out) {
  __shared__ bf16_t bpanel[64 * 32];           // 4 KB weight panel

  const int lane = threadIdx.x & 31;
  const int wave = threadIdx.x >> 5;
  const int ln = lane & 15;
  const int hi = (lane >> 4) & 1;

  const int cb_blocks = Dc / 64;               // 32
  const int rblk = blockIdx.x / cb_blocks;     // over B*L/128 = 32
  const int cblk = blockIdx.x - rblk * cb_blocks;
  const int row_base = rblk * 128;
  const int col_base = cblk * 64;
  const int row0 = row_base + (wave >> 1) * 32;
  const int col0 = (wave & 1) * 32;            // relative to col_base

  v8f acc[2][2] = {};

  for (int kk = 0; kk < Dc; kk += 32) {
#if HAVE_TDM
    if (wave == 0) {
      tdm_load_2d_bf16((unsigned)(size_t)&bpanel[0],
                       wo + (size_t)col_base * Dc + kk,
                       /*tile_d0=*/32, /*tile_d1=*/64,
                       /*tensor_d0=*/Dc, /*tensor_d1=*/Dc,
                       /*stride_d0=*/Dc);
#if __has_builtin(__builtin_amdgcn_s_wait_tensorcnt)
      __builtin_amdgcn_s_wait_tensorcnt(0);
#endif
    }
    __syncthreads();
#else
    // Fallback: cooperative global->LDS copy of the 64x32 panel.
    for (int i = threadIdx.x; i < 64 * 4; i += 256) {
      const int r = i >> 2, c4 = (i & 3) * 8;
      *(bf16x8*)&bpanel[r * 32 + c4] =
          *(const bf16x8*)(wo + (size_t)(col_base + r) * Dc + kk + c4);
    }
    __syncthreads();
#endif
    bf16x16 a0 = load_a_frag(ab, row0 + ln, Dc, kk, lane);
    bf16x16 a1 = load_a_frag(ab, row0 + 16 + ln, Dc, kk, lane);
#pragma unroll
    for (int ct = 0; ct < 2; ++ct) {
      // B fragment from LDS: row = local col, 16 contiguous k per lane.
      bf16x16 bw = *(const bf16x16*)&bpanel[(col0 + ct * 16 + ln) * 32 + hi * 16];
      acc[0][ct] = wmma_bf16(a0, bw, acc[0][ct]);
      acc[1][ct] = wmma_bf16(a1, bw, acc[1][ct]);
    }
    __syncthreads();
  }

#pragma unroll
  for (int rt = 0; rt < 2; ++rt) {
#pragma unroll
    for (int ct = 0; ct < 2; ++ct) {
#pragma unroll
      for (int v = 0; v < 8; ++v) {
        const int row = row0 + rt * 16 + v + (hi << 3);
        const int col = col_base + col0 + ct * 16 + ln;
        out[(size_t)row * Dc + col] = acc[rt][ct][v];
      }
    }
  }
}

// ---------------------------------------------------------------------------
// Host launcher
// ---------------------------------------------------------------------------
extern "C" void kernel_launch(void* const* d_in, const int* in_sizes, int n_in,
                              void* d_out, int out_size, void* d_ws, size_t ws_size,
                              hipStream_t stream) {
  const float* x    = (const float*)d_in[0];
  const float* fcos = (const float*)d_in[1];
  const float* fsin = (const float*)d_in[2];
  const float* Wq   = (const float*)d_in[3];
  const float* Wk   = (const float*)d_in[4];
  const float* Wv   = (const float*)d_in[5];
  const float* Wo   = (const float*)d_in[6];

  const size_t nX = (size_t)Bc * Lc * Dc;   // 8,388,608
  const size_t nW = (size_t)Dc * Dc;        // 4,194,304

  char* ws = (char*)d_ws;
  bf16_t* xb   = (bf16_t*)ws;  ws += nX * sizeof(bf16_t);
  bf16_t* wqb  = (bf16_t*)ws;  ws += nW * sizeof(bf16_t);
  bf16_t* wkb  = (bf16_t*)ws;  ws += nW * sizeof(bf16_t);
  bf16_t* wvb  = (bf16_t*)ws;  ws += nW * sizeof(bf16_t);
  bf16_t* wob  = (bf16_t*)ws;  ws += nW * sizeof(bf16_t);
  bf16_t* Qb   = (bf16_t*)ws;  ws += nX * sizeof(bf16_t);
  bf16_t* Kb   = (bf16_t*)ws;  ws += nX * sizeof(bf16_t);
  bf16_t* Vt   = (bf16_t*)ws;  ws += nX * sizeof(bf16_t);
  bf16_t* attn = (bf16_t*)ws;  ws += nX * sizeof(bf16_t);

  // Stage 0: bf16 conversion (x4 vectorized).
  cvt_f32_bf16_v4<<<(int)(nX / 4 / 256), 256, 0, stream>>>((const float4*)x,
                                                           (bf16x4*)xb, (int)(nX / 4));
  cvt_f32_bf16_v4<<<(int)(nW / 4 / 256), 256, 0, stream>>>((const float4*)Wq,
                                                           (bf16x4*)wqb, (int)(nW / 4));
  cvt_f32_bf16_v4<<<(int)(nW / 4 / 256), 256, 0, stream>>>((const float4*)Wk,
                                                           (bf16x4*)wkb, (int)(nW / 4));
  cvt_f32_bf16_v4<<<(int)(nW / 4 / 256), 256, 0, stream>>>((const float4*)Wv,
                                                           (bf16x4*)wvb, (int)(nW / 4));
  cvt_f32_bf16_v4<<<(int)(nW / 4 / 256), 256, 0, stream>>>((const float4*)Wo,
                                                           (bf16x4*)wob, (int)(nW / 4));

  // Stage 1: QKV + RoPE. (B*L/32)*(D/32) = 128*64 = 8192 waves, 8 per block.
  qkv_rope_kernel<<<8192 / 8, 256, 0, stream>>>(xb, wqb, wkb, wvb, fcos, fsin,
                                                Qb, Kb, Vt);

  // Stage 2: attention. B*H*(L/16) = 4096 waves, 4 waves/block.
  attn_kernel<<<4096 / ATT_WAVES, 32 * ATT_WAVES, 0, stream>>>(Qb, Kb, Vt, attn);

  // Stage 3: output projection -> f32 d_out. (B*L/128)*(D/64) = 1024 blocks.
  proj_kernel<<<1024, 256, 0, stream>>>(attn, wob, (float*)d_out);
}
